// Net_30820685316845
// MI455X (gfx1250) — matrix operand order repaired and checked by reading, not compiled, not run on hardware
//
#include <hip/hip_runtime.h>
#include <hip/hip_bf16.h>

typedef __attribute__((ext_vector_type(16))) _Float16 v16h;
typedef __attribute__((ext_vector_type(8)))  _Float16 v8h;
typedef __attribute__((ext_vector_type(4)))  _Float16 v4h;
typedef __attribute__((ext_vector_type(8)))  float    v8f;

#define N_NODES 20000
#define N_EDGES 320000
#define HID 128

// ---------------- WMMA fragment helpers (gfx1250, wave32) ----------------

__device__ __forceinline__ v16h cat16(v8h lo, v8h hi) {
    v16h a;
#pragma unroll
    for (int i = 0; i < 8; ++i) { a[i] = lo[i]; a[8 + i] = hi[i]; }
    return a;
}

// A-fragment (16xK f16, row-major [16 x stride] halves in LDS) for one K=32 chunk.
// ISA layout: lanes 0-15 hold M=lane, K = kc*32 + {0..7, 16..23};
//             lanes 16-31 hold M=lane-16, K = kc*32 + {8..15, 24..31}.
// -> per lane: two contiguous 8-half (16B) runs => two ds_load_b128.
__device__ __forceinline__ v16h load_a_lds(const _Float16* lds, int stride, int kc) {
    const int lane = threadIdx.x & 31;
    const int M = lane & 15;
    const int d = (lane >> 4) << 3;                 // 0 or 8
    const _Float16* p = lds + M * stride + kc * 32 + d;
    v8h lo = *(const v8h*)p;                        // K offsets d..d+7
    v8h hi = *(const v8h*)(p + 16);                 // K offsets d+16..d+23
    return cat16(lo, hi);
}

// B-fragment from pre-packed global blocks: each (kc,nt) block is
// 32 lanes x 16 halves contiguous (1KB), lane-major.
__device__ __forceinline__ v16h load_b_pk(const _Float16* blk) {
    const int lane = threadIdx.x & 31;
    const _Float16* p = blk + lane * 16;
    v8h lo = *(const v8h*)p;
    v8h hi = *(const v8h*)(p + 8);
    return cat16(lo, hi);
}

__device__ __forceinline__ v8f wmma_f16(v16h a, v16h b, v8f c) {
    return __builtin_amdgcn_wmma_f32_16x16x32_f16(false, a, false, b, (short)0, c,
                                                  false, false);
}

// ---------------- Weight packing: f32 [K,N] -> B-fragment blocks ----------------
// out layout: [KC][NT][lane(32)][e(16)] halves; B ISA map: K = kc*32 + 8*(lane/16) + (e<8 ? e : 8+e),
// N = nt*16 + lane%16. Zero-pads K >= Kreal (used for the 16->32 pad of ea_w1).
__global__ void pack_w_kernel(const float* __restrict__ W, _Float16* __restrict__ out,
                              int Kreal, int N, int KC, int NT) {
    int idx = blockIdx.x * blockDim.x + threadIdx.x;
    int total = KC * NT * 512;
    if (idx >= total) return;
    int e    = idx & 15;
    int lane = (idx >> 4) & 31;
    int rest = idx >> 9;
    int nt   = rest % NT;
    int kc   = rest / NT;
    int k = kc * 32 + ((lane >> 4) << 3) + (e < 8 ? e : e + 8);
    int n = nt * 16 + (lane & 15);
    float v = (k < Kreal) ? W[(size_t)k * N + n] : 0.f;
    out[idx] = (_Float16)v;
}

// ---------------- embed_x: h = relu(emb[x]) @ ex_w + b  (pre-LN) ----------------
__global__ void __launch_bounds__(32) embed_x_kernel(
    const int* __restrict__ xidx, const float* __restrict__ emb,
    const _Float16* __restrict__ wpk, const float* __restrict__ bias,
    float* __restrict__ hout) {
    __shared__ _Float16 arow[16 * HID];
    const int lane = threadIdx.x;
    const int n0 = blockIdx.x * 16;
    for (int q = lane; q < 16 * 32; q += 32) {      // float4-granular gather
        int m = q >> 5, j = (q & 31) * 4;
        float4 v = *(const float4*)(emb + (size_t)xidx[n0 + m] * HID + j);
        v4h h;
        h[0] = (_Float16)(v.x > 0.f ? v.x : 0.f);
        h[1] = (_Float16)(v.y > 0.f ? v.y : 0.f);
        h[2] = (_Float16)(v.z > 0.f ? v.z : 0.f);
        h[3] = (_Float16)(v.w > 0.f ? v.w : 0.f);
        *(v4h*)(arow + m * HID + j) = h;
    }
    __syncthreads();
    v8f acc[8] = {};
#pragma unroll
    for (int kc = 0; kc < 4; ++kc) {
        v16h a = load_a_lds(arow, HID, kc);
#pragma unroll
        for (int nt = 0; nt < 8; ++nt)
            acc[nt] = wmma_f16(a, load_b_pk(wpk + (kc * 8 + nt) * 512), acc[nt]);
    }
    const int Mb = (lane >> 4) << 3, nl = lane & 15;
#pragma unroll
    for (int nt = 0; nt < 8; ++nt) {
        int n = nt * 16 + nl;
        float bv = bias[n];
#pragma unroll
        for (int i = 0; i < 8; ++i)
            hout[(size_t)(n0 + Mb + i) * HID + n] = acc[nt][i] + bv;
    }
}

// ---------------- embed_adj: relu(ea@w1+b1)@w2+b2  (pre-LN) ----------------
__global__ void __launch_bounds__(32) embed_adj_kernel(
    const float* __restrict__ ea, const _Float16* __restrict__ w1pk,
    const float* __restrict__ b1, const _Float16* __restrict__ w2pk,
    const float* __restrict__ b2, float* __restrict__ eout) {
    __shared__ _Float16 ain[16 * 32];
    __shared__ _Float16 hid[16 * HID];
    const int lane = threadIdx.x;
    const int e0 = blockIdx.x * 16;
    for (int q = lane; q < 16 * 4; q += 32) {       // 16 rows x 16 cols, float4
        int m = q >> 2, j = (q & 3) * 4;
        float4 v = *(const float4*)(ea + (size_t)(e0 + m) * 16 + j);
        v4h h; h[0] = (_Float16)v.x; h[1] = (_Float16)v.y;
        h[2] = (_Float16)v.z; h[3] = (_Float16)v.w;
        *(v4h*)(ain + m * 32 + j) = h;
        v4h z = {};                                  // zero-pad K 16->32
        *(v4h*)(ain + m * 32 + 16 + j) = z;
    }
    __syncthreads();
    const int Mb = (lane >> 4) << 3, nl = lane & 15;
    {
        v8f acc[8] = {};
        v16h a = load_a_lds(ain, 32, 0);
#pragma unroll
        for (int nt = 0; nt < 8; ++nt)
            acc[nt] = wmma_f16(a, load_b_pk(w1pk + nt * 512), acc[nt]);
#pragma unroll
        for (int nt = 0; nt < 8; ++nt) {
            int n = nt * 16 + nl;
            float bv = b1[n];
#pragma unroll
            for (int i = 0; i < 8; ++i) {
                float v = acc[nt][i] + bv;
                hid[(Mb + i) * HID + n] = (_Float16)(v > 0.f ? v : 0.f);
            }
        }
    }
    __syncthreads();
    v8f acc[8] = {};
#pragma unroll
    for (int kc = 0; kc < 4; ++kc) {
        v16h a = load_a_lds(hid, HID, kc);
#pragma unroll
        for (int nt = 0; nt < 8; ++nt)
            acc[nt] = wmma_f16(a, load_b_pk(w2pk + (kc * 8 + nt) * 512), acc[nt]);
    }
#pragma unroll
    for (int nt = 0; nt < 8; ++nt) {
        int n = nt * 16 + nl;
        float bv = b2[n];
#pragma unroll
        for (int i = 0; i < 8; ++i)
            eout[(size_t)(e0 + Mb + i) * HID + n] = acc[nt][i] + bv;
    }
}

// ---------------- graph conv messages + scatter-add (M=32 per wave) ----------------
// m = [x[row] | x[col] | e] (32x384) -> relu(m@w1+b1) (32x256) -> @w2+b2 (32x128)
// Two A-fragments share every B-fragment load (B reuse x2 vs 16-row tiles).
__global__ void __launch_bounds__(32) gc_msg_kernel(
    const float* __restrict__ x, const float* __restrict__ e,
    const int* __restrict__ row, const int* __restrict__ col,
    const _Float16* __restrict__ w1pk, const float* __restrict__ b1,
    const _Float16* __restrict__ w2pk, const float* __restrict__ b2,
    float* __restrict__ msg, float* __restrict__ agg) {
    __shared__ _Float16 cat[32 * 384];   // 24 KB
    __shared__ _Float16 hid[32 * 256];   // 16 KB
    const int lane = threadIdx.x;
    const int e0 = blockIdx.x * 32;
    // gather: float4 global loads -> packed 4xf16 LDS stores (ds_store_b64)
    for (int q = lane; q < 32 * 96; q += 32) {
        int m = q / 96, j = (q % 96) * 4;
        int em = e0 + m;
        float4 v;
        if (j < 128)      v = *(const float4*)(x + (size_t)row[em] * HID + j);
        else if (j < 256) v = *(const float4*)(x + (size_t)col[em] * HID + (j - 128));
        else              v = *(const float4*)(e + (size_t)em * HID + (j - 256));
        v4h h; h[0] = (_Float16)v.x; h[1] = (_Float16)v.y;
        h[2] = (_Float16)v.z; h[3] = (_Float16)v.w;
        *(v4h*)(cat + m * 384 + j) = h;
    }
    __syncthreads();
    const int Mb = (lane >> 4) << 3, nl = lane & 15;
    // GEMM1: K=384 (12 chunks), N=256 in 4 passes of 4 column tiles
#pragma unroll
    for (int np = 0; np < 4; ++np) {
        v8f acc[2][4] = {};
        for (int kc = 0; kc < 12; ++kc) {
            v16h a0 = load_a_lds(cat, 384, kc);
            v16h a1 = load_a_lds(cat + 16 * 384, 384, kc);
            if (kc < 11)
                __builtin_prefetch(w1pk + ((kc + 1) * 16 + np * 4) * 512, 0, 0);
#pragma unroll
            for (int nt = 0; nt < 4; ++nt) {
                v16h b = load_b_pk(w1pk + (kc * 16 + np * 4 + nt) * 512);
                acc[0][nt] = wmma_f16(a0, b, acc[0][nt]);
                acc[1][nt] = wmma_f16(a1, b, acc[1][nt]);
            }
        }
#pragma unroll
        for (int t = 0; t < 2; ++t)
#pragma unroll
            for (int nt = 0; nt < 4; ++nt) {
                int n = (np * 4 + nt) * 16 + nl;
                float bv = b1[n];
#pragma unroll
                for (int i = 0; i < 8; ++i) {
                    float v = acc[t][nt][i] + bv;
                    hid[(t * 16 + Mb + i) * 256 + n] = (_Float16)(v > 0.f ? v : 0.f);
                }
            }
    }
    __syncthreads();
    // GEMM2: K=256 (8 chunks), N=128 in 2 passes of 4 column tiles
#pragma unroll
    for (int np = 0; np < 2; ++np) {
        v8f acc[2][4] = {};
#pragma unroll
        for (int kc = 0; kc < 8; ++kc) {
            v16h a0 = load_a_lds(hid, 256, kc);
            v16h a1 = load_a_lds(hid + 16 * 256, 256, kc);
#pragma unroll
            for (int nt = 0; nt < 4; ++nt) {
                v16h b = load_b_pk(w2pk + (kc * 8 + np * 4 + nt) * 512);
                acc[0][nt] = wmma_f16(a0, b, acc[0][nt]);
                acc[1][nt] = wmma_f16(a1, b, acc[1][nt]);
            }
        }
#pragma unroll
        for (int t = 0; t < 2; ++t)
#pragma unroll
            for (int nt = 0; nt < 4; ++nt) {
                int n = (np * 4 + nt) * 16 + nl;
                float bv = b2[n];
#pragma unroll
                for (int i = 0; i < 8; ++i) {
                    int em = e0 + t * 16 + Mb + i;
                    float v = acc[t][nt][i] + bv;
                    msg[(size_t)em * HID + n] = v;
                    atomicAdd(&agg[(size_t)row[em] * HID + n], v);  // segment_sum
                }
            }
    }
}

// ---------------- LayerNorm (+ optional residual): out = res + LN(in)*g + b ----------------
// One wave per row of 128; 8 rows per 256-thread block (wave32 shuffles).
__global__ void ln_res_kernel(const float* __restrict__ in, const float* __restrict__ res,
                              const float* __restrict__ g, const float* __restrict__ b,
                              float* __restrict__ out, int rows) {
    const int wave = threadIdx.x >> 5, lane = threadIdx.x & 31;
    const int r = blockIdx.x * 8 + wave;
    if (r >= rows) return;
    const size_t base = (size_t)r * HID + lane * 4;
    const float4 v = *(const float4*)(in + base);
    float s = v.x + v.y + v.z + v.w;
#pragma unroll
    for (int o = 16; o; o >>= 1) s += __shfl_xor(s, o, 32);
    float mu = s * (1.f / 128.f);
    float dx = v.x - mu, dy = v.y - mu, dz = v.z - mu, dw = v.w - mu;
    float q = dx * dx + dy * dy + dz * dz + dw * dw;
#pragma unroll
    for (int o = 16; o; o >>= 1) q += __shfl_xor(q, o, 32);
    float rstd = rsqrtf(q * (1.f / 128.f) + 1e-5f);
    const int j = lane * 4;
    const float4 gv = *(const float4*)(g + j);
    const float4 bv = *(const float4*)(b + j);
    float4 o4;
    o4.x = dx * rstd * gv.x + bv.x;
    o4.y = dy * rstd * gv.y + bv.y;
    o4.z = dz * rstd * gv.z + bv.z;
    o4.w = dw * rstd * gv.w + bv.w;
    if (res) {
        const float4 rv = *(const float4*)(res + base);
        o4.x += rv.x; o4.y += rv.y; o4.z += rv.z; o4.w += rv.w;
    }
    *(float4*)(out + base) = o4;
}

__global__ void relu_inplace_kernel(float* __restrict__ p, int n) {
    int i = blockIdx.x * blockDim.x + threadIdx.x;
    if (i < n) { float v = p[i]; p[i] = v > 0.f ? v : 0.f; }
}

__global__ void zero_kernel(float* __restrict__ p, int n) {
    int i = blockIdx.x * blockDim.x + threadIdx.x;
    if (i < n) p[i] = 0.f;
}

// ---------------- output head: [N,128]@[128,20]+b ----------------
__global__ void out_head_kernel(const float* __restrict__ x, const float* __restrict__ w,
                                const float* __restrict__ b, float* __restrict__ out) {
    int idx = blockIdx.x * blockDim.x + threadIdx.x;
    if (idx >= N_NODES * 20) return;
    int n = idx / 20, o = idx % 20;
    float s = b[o];
    const float* xr = x + (size_t)n * HID;
#pragma unroll 4
    for (int k = 0; k < HID; ++k) s += xr[k] * w[k * 20 + o];
    out[idx] = s;
}

// ---------------- driver ----------------
extern "C" void kernel_launch(void* const* d_in, const int* in_sizes, int n_in,
                              void* d_out, int out_size, void* d_ws, size_t ws_size,
                              hipStream_t stream) {
    (void)in_sizes; (void)n_in; (void)out_size; (void)ws_size;

    const int*   x_idx = (const int*)d_in[0];
    const int*   row   = (const int*)d_in[1];
    const int*   col   = row + N_EDGES;
    const float* eattr = (const float*)d_in[2];
    const float* emb   = (const float*)d_in[3];
    const float* ex_w  = (const float*)d_in[4];
    const float* ex_b  = (const float*)d_in[5];
    const float* ex_g  = (const float*)d_in[6];
    const float* ex_be = (const float*)d_in[7];
    const float* ea_w1 = (const float*)d_in[8];
    const float* ea_b1 = (const float*)d_in[9];
    const float* ea_w2 = (const float*)d_in[10];
    const float* ea_b2 = (const float*)d_in[11];
    const float* ea_g  = (const float*)d_in[12];
    const float* ea_be = (const float*)d_in[13];
    const float* w_out = (const float*)d_in[46];
    const float* b_out = (const float*)d_in[47];

    // workspace layout (bytes)
    char* ws = (char*)d_ws;
    const size_t SZ_N = (size_t)N_NODES * HID * sizeof(float);   // 10.24 MB
    const size_t SZ_E = (size_t)N_EDGES * HID * sizeof(float);   // 163.84 MB
    float* x_buf = (float*)(ws);
    float* tmp_n = (float*)(ws + SZ_N);                          // embed h / agg
    float* e_buf = (float*)(ws + 2 * SZ_N);
    float* msg   = (float*)(ws + 2 * SZ_N + SZ_E);
    _Float16* pk = (_Float16*)(ws + 2 * SZ_N + 2 * SZ_E);
    _Float16* p_exw  = pk;                                       // 4*8*512
    _Float16* p_eaw1 = p_exw + 16384;                            // 1*8*512
    _Float16* p_eaw2 = p_eaw1 + 4096;                            // 4*8*512
    _Float16* p_gc   = p_eaw2 + 16384;                           // 4 * (12*16 + 8*8)*512

    // ---- pack weights to B-fragment layout (f32 -> f16) ----
    auto packs = [&](const float* W, _Float16* dst, int K, int N, int KC, int NT) {
        int total = KC * NT * 512;
        pack_w_kernel<<<(total + 255) / 256, 256, 0, stream>>>(W, dst, K, N, KC, NT);
    };
    packs(ex_w,  p_exw,  128, 128, 4, 8);
    packs(ea_w1, p_eaw1,  16, 128, 1, 8);
    packs(ea_w2, p_eaw2, 128, 128, 4, 8);
    for (int l = 0; l < 4; ++l) {
        const float* w1 = (const float*)d_in[14 + l * 8 + 0];
        const float* w2 = (const float*)d_in[14 + l * 8 + 2];
        _Float16* pw1 = p_gc + (size_t)l * (12 * 16 + 8 * 8) * 512;
        _Float16* pw2 = pw1 + 12 * 16 * 512;
        packs(w1, pw1, 384, 256, 12, 16);
        packs(w2, pw2, 256, 128, 8, 8);
    }

    // ---- embed_x: x = LN(relu(emb[x]) @ ex_w + b) ----
    embed_x_kernel<<<N_NODES / 16, 32, 0, stream>>>(x_idx, emb, p_exw, ex_b, tmp_n);
    ln_res_kernel<<<N_NODES / 8, 256, 0, stream>>>(tmp_n, nullptr, ex_g, ex_be, x_buf,
                                                   N_NODES);
    // ---- embed_adj: e = LN(mlp2(edge_attr)) ----
    embed_adj_kernel<<<N_EDGES / 16, 32, 0, stream>>>(eattr, p_eaw1, ea_b1, p_eaw2,
                                                      ea_b2, msg);
    ln_res_kernel<<<N_EDGES / 8, 256, 0, stream>>>(msg, nullptr, ea_g, ea_be, e_buf,
                                                   N_EDGES);

    // ---- 4 graph-conv layers ----
    for (int l = 0; l < 4; ++l) {
        const float* b1 = (const float*)d_in[14 + l * 8 + 1];
        const float* b2 = (const float*)d_in[14 + l * 8 + 3];
        const float* gx = (const float*)d_in[14 + l * 8 + 4];
        const float* bx = (const float*)d_in[14 + l * 8 + 5];
        const float* ge = (const float*)d_in[14 + l * 8 + 6];
        const float* be = (const float*)d_in[14 + l * 8 + 7];
        _Float16* pw1 = p_gc + (size_t)l * (12 * 16 + 8 * 8) * 512;
        _Float16* pw2 = pw1 + 12 * 16 * 512;

        if (l > 0) {
            relu_inplace_kernel<<<(N_NODES * HID + 255) / 256, 256, 0, stream>>>(
                x_buf, N_NODES * HID);
            relu_inplace_kernel<<<(N_EDGES * HID + 255) / 256, 256, 0, stream>>>(
                e_buf, N_EDGES * HID);
        }
        zero_kernel<<<(N_NODES * HID + 255) / 256, 256, 0, stream>>>(tmp_n,
                                                                     N_NODES * HID);
        gc_msg_kernel<<<N_EDGES / 32, 32, 0, stream>>>(x_buf, e_buf, row, col, pw1, b1,
                                                       pw2, b2, msg, tmp_n);
        // x = x + LN(agg); e = e + LN(msg)
        ln_res_kernel<<<N_NODES / 8, 256, 0, stream>>>(tmp_n, x_buf, gx, bx, x_buf,
                                                       N_NODES);
        ln_res_kernel<<<N_EDGES / 8, 256, 0, stream>>>(msg, e_buf, ge, be, e_buf,
                                                       N_EDGES);
    }

    // ---- output head ----
    out_head_kernel<<<(N_NODES * 20 + 255) / 256, 256, 0, stream>>>(x_buf, w_out, b_out,
                                                                    (float*)d_out);
}
